// MLSTM_18983755448554
// MI455X (gfx1250) — matrix operand order repaired and checked by reading, not compiled, not run on hardware
//
#include <hip/hip_runtime.h>
#include <hip/hip_bf16.h>

#ifndef __has_builtin
#define __has_builtin(x) 0
#endif

// ---------------------------------------------------------------------------
// mLSTM for MI455X (gfx1250, wave32, WMMA bf16, TDM).
//
// Fast path (ws_size >= ~676MB):
//   Phase A: full-chip WMMA GEMM precomputes P = x @ [W_ih; mx_w]^T for all
//            T*B rows (343 GFLOP at chip-wide WMMA rate), stored bf16 in a
//            C/D-register-friendly layout [tb][2560][16rows].
//   Phase B: persistent recurrence (16 blocks x 32 waves), per step only
//            h@mh_w^T + m@W_hh^T (80 WMMAs/wave/step). Each step's 80KB
//            activation slab is DMA'd into LDS by the Tensor Data Mover
//            (tensor_load_to_lds, double-buffered, s_wait_tensorcnt).
// Fallback path: fused persistent kernel (round-1).
// ---------------------------------------------------------------------------

#define Tn 512
#define Bn 256
#define In 512
#define Hn 512
#define SLAB_DW (2560 * 16 * 2 / 4)   /* 20480 dwords = 80KB per step-tile */

typedef __attribute__((ext_vector_type(16))) __bf16 bf16x16;
typedef __attribute__((ext_vector_type(8)))  float  v8f;
typedef __attribute__((ext_vector_type(4)))  unsigned int u32x4;
typedef __attribute__((ext_vector_type(8)))  unsigned int u32x8;
typedef __attribute__((ext_vector_type(4)))  int i32x4;
typedef __attribute__((ext_vector_type(8)))  int i32x8;

__device__ __forceinline__ unsigned short f2bf(float f) {
    unsigned u = __builtin_bit_cast(unsigned, f);
    u += 0x7FFFu + ((u >> 16) & 1u);           // round-to-nearest-even
    return (unsigned short)(u >> 16);
}

__device__ __forceinline__ bf16x16 cat16(u32x4 a, u32x4 b) {
    u32x8 t;
    t[0] = a[0]; t[1] = a[1]; t[2] = a[2]; t[3] = a[3];
    t[4] = b[0]; t[5] = b[1]; t[6] = b[2]; t[7] = b[3];
    return __builtin_bit_cast(bf16x16, t);
}

__device__ __forceinline__ v8f zero8() {
    v8f z = {0.f, 0.f, 0.f, 0.f, 0.f, 0.f, 0.f, 0.f};
    return z;
}

// unpack 8 bf16 (one u32x4) to 8 floats, in element order
__device__ __forceinline__ void unpack8(u32x4 v, float* f) {
#pragma unroll
    for (int i = 0; i < 4; ++i) {
        unsigned w = v[i];
        f[2 * i]     = __builtin_bit_cast(float, w << 16);
        f[2 * i + 1] = __builtin_bit_cast(float, w & 0xFFFF0000u);
    }
}

// A-matrix tile (16x32 bf16) from LDS, row-major [16][512].
__device__ __forceinline__ bf16x16 load_a_lds(const unsigned short* base,
                                              int m, int k0, int hi) {
    const u32x4* p1 = (const u32x4*)(base + m * Hn + k0 + hi * 16);
    const u32x4* p2 = (const u32x4*)(base + m * Hn + k0 + hi * 16 + 8);
    return cat16(*p1, *p2);
}

// B-matrix tile (32x16 bf16) from global row-major weight [N][512].
__device__ __forceinline__ bf16x16 load_b_glb(const unsigned short* __restrict__ w,
                                              int row, int k0, int hi) {
    const u32x4* q = (const u32x4*)(w + (size_t)row * 512 + k0 + hi * 16);
    return cat16(q[0], q[1]);
}

__device__ __forceinline__ bf16x16 load_b_ptr(const unsigned short* wrow,
                                              int k0, int hi) {
    const u32x4* q = (const u32x4*)(wrow + k0 + hi * 16);
    return cat16(q[0], q[1]);
}

__device__ __forceinline__ v8f wmma_bf16(bf16x16 a, bf16x16 b, v8f c) {
    return __builtin_amdgcn_wmma_f32_16x16x32_bf16(false, a, false, b,
                                                   (short)0, c, false, false);
}

__device__ __forceinline__ float sigf(float x) { return 1.0f / (1.0f + __expf(-x)); }
__device__ __forceinline__ float tanh_apx(float x) {
    float e = __expf(-2.0f * x);
    return (1.0f - e) / (1.0f + e);
}

// ---- Tensor Data Mover: 1-row tile of `ndw` dwords, global -> LDS --------
#if __has_builtin(__builtin_amdgcn_tensor_load_to_lds)
#define HAVE_TDM 1
#else
#define HAVE_TDM 0
#endif

__device__ __forceinline__ void tdm_load_slab(unsigned lds_byte,
                                              unsigned long long gaddr,
                                              unsigned ndw) {
#if HAVE_TDM
    u32x4 g0;
    g0[0] = 1u;                                   // count=1, user mode
    g0[1] = lds_byte;                             // D#.lds_addr
    g0[2] = (unsigned)(gaddr & 0xFFFFFFFFull);    // D#.global_addr[31:0]
    g0[3] = ((unsigned)(gaddr >> 32) & 0x1FFFFFFu) | 0x80000000u; // addr[56:32] | type=2
    i32x8 g1;
    g1[0] = 0x20000;                              // data_size = 4B
    g1[1] = (int)((ndw & 0xFFFFu) << 16);         // tensor_dim0[15:0]
    g1[2] = (int)((ndw >> 16) | (1u << 16));      // tensor_dim0[31:16] | tensor_dim1=1
    g1[3] = (int)((ndw & 0xFFFFu) << 16);         // tile_dim0 = ndw
    g1[4] = 1;                                    // tile_dim1 = 1
    g1[5] = (int)ndw;                             // tensor_dim0_stride
    g1[6] = 0; g1[7] = 0;
    i32x4 z4 = {0, 0, 0, 0};
#if __clang_major__ >= 23
    i32x8 z8 = {0, 0, 0, 0, 0, 0, 0, 0};
    __builtin_amdgcn_tensor_load_to_lds(g0, g1, z4, z4, z8, 0);
#else
    __builtin_amdgcn_tensor_load_to_lds(g0, g1, z4, z4, 0);
#endif
#endif
}

template <int N>
__device__ __forceinline__ void tdm_wait() {
#if __has_builtin(__builtin_amdgcn_s_wait_tensorcnt)
    __builtin_amdgcn_s_wait_tensorcnt(N);   // immediate required
#endif
}

// ===========================================================================
// fp32 -> bf16 weight conversion
// ===========================================================================
__global__ void mlstm_cvt_bf16(const float* __restrict__ src,
                               unsigned short* __restrict__ dst, int n) {
    int i = blockIdx.x * blockDim.x + threadIdx.x;
    int stride = gridDim.x * blockDim.x;
    for (; i < n; i += stride) dst[i] = f2bf(src[i]);
}

// ===========================================================================
// Phase A: P[tb][col 0..2559][row 0..15] = (x @ [W_ih; mx_w]^T) in bf16
//   grid = (20 colgroups, 8192 row-tiles), 256 threads = 8 waves.
// ===========================================================================
__global__ __launch_bounds__(256)
void mlstm_precompute(const float* __restrict__ x,
                      const unsigned short* __restrict__ Wih,
                      const unsigned short* __restrict__ mxw,
                      unsigned short* __restrict__ P) {
    __shared__ unsigned short xS[16 * In];

    const int tid  = threadIdx.x;
    const int lane = tid & 31;
    const int wave = tid >> 5;
    const int hi   = lane >> 4;
    const int l15  = lane & 15;
    const size_t mt = blockIdx.y;                 // row tile: t*16 + btile

    // load + convert 16 rows of x
    const float* xb = x + mt * 16 * In;
#pragma unroll
    for (int it = 0; it < 8; ++it) {
        int e4  = tid + it * 256;
        int row = e4 >> 7;
        int col = (e4 & 127) * 4;
        float4 v = *(const float4*)(xb + row * In + col);
        unsigned long long pk =
            (unsigned long long)f2bf(v.x) |
            ((unsigned long long)f2bf(v.y) << 16) |
            ((unsigned long long)f2bf(v.z) << 32) |
            ((unsigned long long)f2bf(v.w) << 48);
        *(unsigned long long*)(xS + row * In + col) = pk;
    }
    __syncthreads();

    const int n = blockIdx.x * 128 + wave * 16 + l15;   // output col [0,2560)
    const unsigned short* wrow =
        (n < 2048) ? (Wih + (size_t)n * 512) : (mxw + (size_t)(n - 2048) * 512);

    v8f acc = zero8();
#pragma unroll 4
    for (int k0 = 0; k0 < In; k0 += 32) {
        bf16x16 a = load_a_lds(xS, l15, k0, hi);
        acc = wmma_bf16(a, load_b_ptr(wrow, k0, hi), acc);
    }

    // pack 8 rows (hi*8 .. hi*8+7) for column n into one b128 store
    u32x4 o;
#pragma unroll
    for (int i = 0; i < 4; ++i)
        o[i] = (unsigned)f2bf(acc[2 * i]) | ((unsigned)f2bf(acc[2 * i + 1]) << 16);
    *(u32x4*)(P + ((size_t)mt * 2560 + n) * 16 + hi * 8) = o;
}

// ===========================================================================
// Phase B: persistent recurrence with TDM-fed activation slabs.
//   16 blocks (batch tiles) x 1024 threads (32 waves, one 16-col tile each).
// ===========================================================================
__global__ __launch_bounds__(1024)
void mlstm_recur_pre(const unsigned short* __restrict__ P,
                     const int* __restrict__ lengths,
                     const float* __restrict__ b_ih,
                     const float* __restrict__ b_hh,
                     const float* __restrict__ mx_b,
                     const float* __restrict__ mh_b,
                     const float* __restrict__ h0,
                     const float* __restrict__ c0,
                     const unsigned short* __restrict__ Whh,
                     const unsigned short* __restrict__ mhw,
                     float* __restrict__ out) {
    __shared__ unsigned short slab[2][2560 * 16];   // 2 x 80KB, TDM dest
    __shared__ unsigned short hS[16 * Hn];
    __shared__ unsigned short mS[16 * Hn];

    const int tid  = threadIdx.x;
    const int lane = tid & 31;
    const int wave = tid >> 5;
    const int hi   = lane >> 4;
    const int l15  = lane & 15;
    const int n    = wave * 16 + l15;
    const int b0   = blockIdx.x * 16;

    float c_acc[8], h_reg[8];
    int   len_r[8];
    {
        const float c0n = c0[n], h0n = h0[n];
#pragma unroll
        for (int r = 0; r < 8; ++r) {
            c_acc[r] = c0n; h_reg[r] = h0n;
            len_r[r] = lengths[b0 + r + 8 * hi];
        }
    }
    const float bi  = b_ih[n]          + b_hh[n];
    const float bff = b_ih[Hn + n]     + b_hh[Hn + n];
    const float bg  = b_ih[2 * Hn + n] + b_hh[2 * Hn + n];
    const float bo  = b_ih[3 * Hn + n] + b_hh[3 * Hn + n];
    const float mxb = mx_b[n];
    const float mhb = mh_b[n];

    for (int e = tid; e < 16 * Hn; e += 1024) hS[e] = f2bf(h0[e & (Hn - 1)]);

    const unsigned long long Pbase = (unsigned long long)(uintptr_t)P;
    const unsigned ldsA = (unsigned)(uintptr_t)(void*)&slab[0][0];
    const unsigned ldsB = (unsigned)(uintptr_t)(void*)&slab[1][0];

#if HAVE_TDM
    if (wave == 0)
        tdm_load_slab(ldsA, Pbase + (unsigned long long)blockIdx.x * 81920ull, SLAB_DW);
#endif
    __syncthreads();

    for (int t = 0; t < Tn; ++t) {
        const unsigned short* xg = &slab[t & 1][0];
#if HAVE_TDM
        if (wave == 0) {
            if (t + 1 < Tn) {
                unsigned long long ga =
                    Pbase + (unsigned long long)((t + 1) * 16 + blockIdx.x) * 81920ull;
                tdm_load_slab((t & 1) ? ldsA : ldsB, ga, SLAB_DW);
                tdm_wait<1>();        // slab[t&1] (issued last step) is done
            } else {
                tdm_wait<0>();
            }
        }
#else
        {   // fallback: cooperative synchronous copy of this step's slab
            const unsigned* src = (const unsigned*)
                (P + ((size_t)t * 16 + blockIdx.x) * 2560 * 16);
            unsigned* dst = (unsigned*)xg;
            for (int e = tid; e < SLAB_DW; e += 1024) dst[e] = src[e];
        }
#endif
        __syncthreads();   // slab[t&1], hS(prev) visible to all

        // ---- stage 1: mp = h @ mh_w^T ; m = xm * (mp + mh_b) ----
        v8f amp = zero8();
#pragma unroll 4
        for (int k0 = 0; k0 < Hn; k0 += 32) {
            bf16x16 ah = load_a_lds(hS, l15, k0, hi);
            amp = wmma_bf16(ah, load_b_glb(mhw, n, k0, hi), amp);
        }
        float xmv[8];
        unpack8(*(const u32x4*)(xg + (size_t)(2048 + n) * 16 + hi * 8), xmv);
#pragma unroll
        for (int r = 0; r < 8; ++r) {
            float mval = (xmv[r] + mxb) * (amp[r] + mhb);
            mS[(r + 8 * hi) * Hn + n] = f2bf(mval);
        }
        __syncthreads();

        // ---- stage 2: gates = xg + m @ W_hh^T ----
        v8f g[4];
        float xv[8];
#pragma unroll
        for (int q = 0; q < 4; ++q) {
            unpack8(*(const u32x4*)(xg + (size_t)(q * Hn + n) * 16 + hi * 8), xv);
#pragma unroll
            for (int r = 0; r < 8; ++r) g[q][r] = xv[r];
        }
#pragma unroll 2
        for (int k0 = 0; k0 < Hn; k0 += 32) {
            bf16x16 am = load_a_lds(mS, l15, k0, hi);
            g[0] = wmma_bf16(am, load_b_glb(Whh, n,          k0, hi), g[0]);
            g[1] = wmma_bf16(am, load_b_glb(Whh, Hn + n,     k0, hi), g[1]);
            g[2] = wmma_bf16(am, load_b_glb(Whh, 2 * Hn + n, k0, hi), g[2]);
            g[3] = wmma_bf16(am, load_b_glb(Whh, 3 * Hn + n, k0, hi), g[3]);
        }
#pragma unroll
        for (int r = 0; r < 8; ++r) {
            float iv = sigf(g[0][r] + bi);
            float fv = sigf(g[1][r] + bff);
            float gv = tanh_apx(g[2][r] + bg);
            float ov = sigf(g[3][r] + bo);
            float cn = fv * c_acc[r] + iv * gv;
            float hn = ov * tanh_apx(cn);
            bool act = (t < len_r[r]);
            c_acc[r] = act ? cn : c_acc[r];
            h_reg[r] = act ? hn : h_reg[r];
            hS[(r + 8 * hi) * Hn + n] = f2bf(h_reg[r]);
        }
        __syncthreads();   // hS/mS reads done; next TDM may overwrite slab[t&1]
    }

#pragma unroll
    for (int r = 0; r < 8; ++r)
        out[(size_t)(b0 + r + 8 * hi) * Hn + n] = c_acc[r];
}

// ===========================================================================
// Fallback: fused persistent kernel (round-1, no big workspace needed)
// ===========================================================================
__global__ __launch_bounds__(1024)
void mlstm_persistent(const float* __restrict__ x,
                      const int* __restrict__ lengths,
                      const float* __restrict__ b_ih,
                      const float* __restrict__ b_hh,
                      const float* __restrict__ mx_b,
                      const float* __restrict__ mh_b,
                      const float* __restrict__ h0,
                      const float* __restrict__ c0,
                      const unsigned short* __restrict__ Wih,
                      const unsigned short* __restrict__ Whh,
                      const unsigned short* __restrict__ mxw,
                      const unsigned short* __restrict__ mhw,
                      float* __restrict__ out) {
    __shared__ unsigned short xS[16 * Hn];
    __shared__ unsigned short hS[16 * Hn];
    __shared__ unsigned short mS[16 * Hn];

    const int tid  = threadIdx.x;
    const int lane = tid & 31;
    const int wave = tid >> 5;
    const int hi   = lane >> 4;
    const int l15  = lane & 15;
    const int n    = wave * 16 + l15;
    const int b0   = blockIdx.x * 16;

    float c_acc[8], h_reg[8];
    int   len_r[8];
    {
        const float c0n = c0[n], h0n = h0[n];
#pragma unroll
        for (int r = 0; r < 8; ++r) {
            c_acc[r] = c0n; h_reg[r] = h0n;
            len_r[r] = lengths[b0 + r + 8 * hi];
        }
    }
    const float bi  = b_ih[n]          + b_hh[n];
    const float bff = b_ih[Hn + n]     + b_hh[Hn + n];
    const float bg  = b_ih[2 * Hn + n] + b_hh[2 * Hn + n];
    const float bo  = b_ih[3 * Hn + n] + b_hh[3 * Hn + n];
    const float mxb = mx_b[n];
    const float mhb = mh_b[n];

    for (int e = tid; e < 16 * Hn; e += 1024) hS[e] = f2bf(h0[e & (Hn - 1)]);
    __syncthreads();

    for (int t = 0; t < Tn; ++t) {
        {
            const float* xb = x + ((size_t)t * Bn + b0) * In;
#pragma unroll
            for (int it = 0; it < 2; ++it) {
                int e4  = tid + it * 1024;
                int row = e4 >> 7;
                int col = (e4 & 127) * 4;
                float4 v = *(const float4*)(xb + row * In + col);
                unsigned long long pk =
                    (unsigned long long)f2bf(v.x) |
                    ((unsigned long long)f2bf(v.y) << 16) |
                    ((unsigned long long)f2bf(v.z) << 32) |
                    ((unsigned long long)f2bf(v.w) << 48);
                *(unsigned long long*)(xS + row * Hn + col) = pk;
            }
            if (t + 1 < Tn)
                __builtin_prefetch((const void*)(xb + (size_t)Bn * In + tid * 8), 0, 1);
        }
        __syncthreads();

        v8f axm = zero8(), amp = zero8();
#pragma unroll 4
        for (int k0 = 0; k0 < Hn; k0 += 32) {
            bf16x16 ax = load_a_lds(xS, l15, k0, hi);
            bf16x16 ah = load_a_lds(hS, l15, k0, hi);
            axm = wmma_bf16(ax, load_b_glb(mxw, n, k0, hi), axm);
            amp = wmma_bf16(ah, load_b_glb(mhw, n, k0, hi), amp);
        }
#pragma unroll
        for (int r = 0; r < 8; ++r) {
            float mval = (axm[r] + mxb) * (amp[r] + mhb);
            mS[(r + 8 * hi) * Hn + n] = f2bf(mval);
        }
        __syncthreads();

        v8f g0 = zero8(), g1 = zero8(), g2 = zero8(), g3 = zero8();
#pragma unroll 2
        for (int k0 = 0; k0 < Hn; k0 += 32) {
            bf16x16 ax = load_a_lds(xS, l15, k0, hi);
            bf16x16 am = load_a_lds(mS, l15, k0, hi);
            g0 = wmma_bf16(ax, load_b_glb(Wih, n,          k0, hi), g0);
            g0 = wmma_bf16(am, load_b_glb(Whh, n,          k0, hi), g0);
            g1 = wmma_bf16(ax, load_b_glb(Wih, Hn + n,     k0, hi), g1);
            g1 = wmma_bf16(am, load_b_glb(Whh, Hn + n,     k0, hi), g1);
            g2 = wmma_bf16(ax, load_b_glb(Wih, 2 * Hn + n, k0, hi), g2);
            g2 = wmma_bf16(am, load_b_glb(Whh, 2 * Hn + n, k0, hi), g2);
            g3 = wmma_bf16(ax, load_b_glb(Wih, 3 * Hn + n, k0, hi), g3);
            g3 = wmma_bf16(am, load_b_glb(Whh, 3 * Hn + n, k0, hi), g3);
        }
#pragma unroll
        for (int r = 0; r < 8; ++r) {
            float iv = sigf(g0[r] + bi);
            float fv = sigf(g1[r] + bff);
            float gv = tanh_apx(g2[r] + bg);
            float ov = sigf(g3[r] + bo);
            float cn = fv * c_acc[r] + iv * gv;
            float hn = ov * tanh_apx(cn);
            bool act = (t < len_r[r]);
            c_acc[r] = act ? cn : c_acc[r];
            h_reg[r] = act ? hn : h_reg[r];
            hS[(r + 8 * hi) * Hn + n] = f2bf(h_reg[r]);
        }
        __syncthreads();
    }

#pragma unroll
    for (int r = 0; r < 8; ++r)
        out[(size_t)(b0 + r + 8 * hi) * Hn + n] = c_acc[r];
}

// ===========================================================================
extern "C" void kernel_launch(void* const* d_in, const int* in_sizes, int n_in,
                              void* d_out, int out_size, void* d_ws, size_t ws_size,
                              hipStream_t stream) {
    const float* x    = (const float*)d_in[0];
    const int*   lens = (const int*)d_in[1];
    const float* W_ih = (const float*)d_in[2];
    const float* W_hh = (const float*)d_in[3];
    const float* b_ih = (const float*)d_in[4];
    const float* b_hh = (const float*)d_in[5];
    const float* mx_w = (const float*)d_in[6];
    const float* mx_b = (const float*)d_in[7];
    const float* mh_w = (const float*)d_in[8];
    const float* mh_b = (const float*)d_in[9];
    const float* h0   = (const float*)d_in[10];
    const float* c0   = (const float*)d_in[11];

    unsigned short* ws     = (unsigned short*)d_ws;
    unsigned short* Wih_bf = ws;                              // 2048*512
    unsigned short* Whh_bf = Wih_bf + 2048 * 512;             // 2048*512
    unsigned short* mxw_bf = Whh_bf + 2048 * 512;             // 512*512
    unsigned short* mhw_bf = mxw_bf + 512 * 512;              // 512*512
    unsigned short* P      = mhw_bf + 512 * 512;              // Tn*16*2560*16

    const size_t wbytes = (size_t)(2 * 2048 * 512 + 2 * 512 * 512) * 2;
    const size_t pbytes = (size_t)Tn * 16 * 2560 * 16 * 2;

    mlstm_cvt_bf16<<<1024, 256, 0, stream>>>(W_ih, Wih_bf, 2048 * 512);
    mlstm_cvt_bf16<<<1024, 256, 0, stream>>>(W_hh, Whh_bf, 2048 * 512);
    mlstm_cvt_bf16<<<512, 256, 0, stream>>>(mx_w, mxw_bf, 512 * 512);
    mlstm_cvt_bf16<<<512, 256, 0, stream>>>(mh_w, mhw_bf, 512 * 512);

    if (ws_size >= wbytes + pbytes) {
        // Phase A: chip-wide GEMM; Phase B: halved-critical-path recurrence
        dim3 gA(20, Tn * 16);
        mlstm_precompute<<<gA, 256, 0, stream>>>(x, Wih_bf, mxw_bf, P);
        mlstm_recur_pre<<<Bn / 16, 1024, 0, stream>>>(
            P, lens, b_ih, b_hh, mx_b, mh_b, h0, c0,
            Whh_bf, mhw_bf, (float*)d_out);
    } else {
        mlstm_persistent<<<Bn / 16, 1024, 0, stream>>>(
            x, lens, b_ih, b_hh, mx_b, mh_b, h0, c0,
            Wih_bf, Whh_bf, mxw_bf, mhw_bf, (float*)d_out);
    }
}
